// PointNet_44598940401961
// MI455X (gfx1250) — matrix operand order repaired and checked by reading, not compiled, not run on hardware
//
#include <hip/hip_runtime.h>

// ---------------------------------------------------------------------------
// PointNet (B=16, N=8192, 3->64->64->64->128->1024, pool, FC 1024->512->40)
// MI455X / gfx1250, wave32, WMMA f32_16x16x32_f16, async global->LDS staging.
// GEMM convention: M = points, N = out-channels, K = in-channels.
// ---------------------------------------------------------------------------

typedef __attribute__((ext_vector_type(16))) _Float16 v16h;
typedef __attribute__((ext_vector_type(8)))  _Float16 v8h;
typedef __attribute__((ext_vector_type(8)))  float    v8f;

#define BN_EPS 1e-5f
#define NB   16      // batch
#define NPT  8192    // points, pass 1
#define KSEL 600     // selected points
#define NPT2 640     // padded pass-2 points (5 tiles of 128)
#define EMB  1024
#define LDS_S 72     // f16 row stride for 64-ch LDS activation tiles

__device__ __forceinline__ v16h cat8(v8h a, v8h b) {
  return __builtin_shufflevector(a, b, 0,1,2,3,4,5,6,7,8,9,10,11,12,13,14,15);
}
__device__ __forceinline__ v8f wmma16(v16h a, v16h b, v8f c) {
  return __builtin_amdgcn_wmma_f32_16x16x32_f16(false, a, false, b, (short)0, c,
                                                false, false);
}
// A (16x32, f16): lane holds row M=lane%16; elems 0..7 = K {k0+8*hi..+7},
// elems 8..15 = K {k0+16+8*hi..+7}   (ISA 7.12.2, 16-bit A 16x32)
__device__ __forceinline__ v16h loadA(const _Float16* base, int row, int stride,
                                      int k0, int hi) {
  const _Float16* p = base + (size_t)row * stride + k0 + hi * 8;
  return cat8(*(const v8h*)p, *(const v8h*)(p + 16));
}
// B (32x16, f16): lane holds col N=lane%16; elems 0..7 = K {k0+8*hi..+7},
// elems 8..15 = K {k0+16+8*hi..+7}. Row `ch` of the [Cout][Cin] weight matrix.
__device__ __forceinline__ v16h loadB(const _Float16* W, int chRow, int stride,
                                      int k0, int hi) {
  const _Float16* p = W + (size_t)chRow * stride + k0 + hi * 8;
  return cat8(*(const v8h*)p, *(const v8h*)(p + 16));
}

// ------------------------------- BN folding --------------------------------
__global__ void fold_bn_f16(const float* __restrict__ W, const float* __restrict__ g,
                            const float* __restrict__ b, const float* __restrict__ m,
                            const float* __restrict__ v, _Float16* __restrict__ Wd,
                            float* __restrict__ bd, int O, int C) {
  int i = blockIdx.x * blockDim.x + threadIdx.x;
  if (i >= O * C) return;
  int o = i / C, c = i % C;
  float s = g[o] * rsqrtf(v[o] + BN_EPS);
  Wd[i] = (_Float16)(W[i] * s);
  if (c == 0) bd[o] = b[o] - m[o] * s;
}
__global__ void fold_w1(const float* __restrict__ W, const float* __restrict__ g,
                        const float* __restrict__ b, const float* __restrict__ m,
                        const float* __restrict__ v, float* __restrict__ Wd,
                        float* __restrict__ bd) {
  int o = threadIdx.x;
  if (o >= 64) return;
  float s = g[o] * rsqrtf(v[o] + BN_EPS);
  Wd[o * 4 + 0] = W[o * 3 + 0] * s;
  Wd[o * 4 + 1] = W[o * 3 + 1] * s;
  Wd[o * 4 + 2] = W[o * 3 + 2] * s;
  Wd[o * 4 + 3] = 0.f;
  bd[o] = b[o] - m[o] * s;
}
__global__ void pad_l2(const float* __restrict__ L2w, _Float16* __restrict__ Wd) {
  int i = blockIdx.x * blockDim.x + threadIdx.x;
  if (i >= 48 * 512) return;
  int o = i / 512;
  Wd[i] = (o < 40) ? (_Float16)L2w[i] : (_Float16)0.f;
}

// -------------------- fused conv1..conv4 (both passes) ---------------------
// Block = 256 threads = 8 waves; tile = 128 points of one batch.
// Layer1 (K=3) in VALU, layers 2..4 via WMMA through LDS (stride 72 f16).
// Layer-4 output is staged in LDS and written out as coalesced b128 stores.
__global__ __launch_bounds__(256) void front_kernel(
    const float* __restrict__ x, const int* __restrict__ sel,
    const float* __restrict__ W1e, const float* __restrict__ b1e,
    const _Float16* __restrict__ W2e, const float* __restrict__ b2e,
    const _Float16* __restrict__ W3e, const float* __restrict__ b3e,
    const _Float16* __restrict__ W4e, const float* __restrict__ b4e,
    _Float16* __restrict__ h4out, int rowsPerBatch, int nValid, int pass) {
  __shared__ __align__(16) _Float16 bufA[128 * LDS_S];
  __shared__ __align__(16) _Float16 bufB[128 * LDS_S];
  __shared__ __align__(16) _Float16 stg[128 * 128];   // layer-4 output tile
  const int b = blockIdx.y;
  const int tileBase = blockIdx.x * 128;
  const int t = threadIdx.x;

  { // layer 1: 3 -> 64, per point in VALU (two threads per point)
    int j = t >> 1;
    int ch0 = (t & 1) * 32;
    int jj = tileBase + j;
    float x0 = 0.f, x1 = 0.f, x2 = 0.f;
    if (jj < nValid) {
      int p = pass ? sel[b * NPT2 + jj] : jj;
      const float* xb = x + (size_t)b * 3 * NPT;
      x0 = xb[p]; x1 = xb[NPT + p]; x2 = xb[2 * NPT + p];
    }
    for (int c = 0; c < 32; ++c) {
      int ch = ch0 + c;
      float s = b1e[ch] + W1e[ch * 4 + 0] * x0 + W1e[ch * 4 + 1] * x1 +
                W1e[ch * 4 + 2] * x2;
      bufA[j * LDS_S + ch] = (_Float16)fmaxf(s, 0.f);
    }
  }
  __syncthreads();

  const int lane = t & 31, w = t >> 5, lrow = lane & 15, hi = lane >> 4;
  const int r0 = w * 16;  // this wave's 16-point sub-tile

  // layers 2 and 3: 64 -> 64   (L2: bufA->bufB, L3: bufB->bufA)
  const _Float16* Win = bufA;
  _Float16* Wout = bufB;
  const _Float16* Wmat = W2e; const float* bv2 = b2e;
  for (int layer = 0; layer < 2; ++layer) {
    for (int c = 0; c < 4; ++c) {
      int ch = c * 16 + lrow;
      v8f acc = {};
      for (int k = 0; k < 2; ++k) {
        v16h a = loadA(Win, r0 + lrow, LDS_S, k * 32, hi);
        v16h bmat = loadB(Wmat, ch, 64, k * 32, hi);
        acc = wmma16(a, bmat, acc);
      }
      float bias = bv2[ch];
      for (int r = 0; r < 8; ++r) {
        int pt = r0 + r + hi * 8;
        Wout[pt * LDS_S + ch] = (_Float16)fmaxf(acc[r] + bias, 0.f);
      }
    }
    __syncthreads();
    Win = bufB; Wout = bufA; Wmat = W3e; bv2 = b3e;
  }

  // layer 4: 64 -> 128, reads bufA (layer-3 output), stages into LDS
  for (int c = 0; c < 8; ++c) {
    int ch = c * 16 + lrow;
    v8f acc = {};
    for (int k = 0; k < 2; ++k) {
      v16h a = loadA(bufA, r0 + lrow, LDS_S, k * 32, hi);
      v16h bmat = loadB(W4e, ch, 64, k * 32, hi);
      acc = wmma16(a, bmat, acc);
    }
    float bias = b4e[ch];
    for (int r = 0; r < 8; ++r) {
      int pt = r0 + r + hi * 8;
      stg[pt * 128 + ch] = (_Float16)fmaxf(acc[r] + bias, 0.f);
    }
  }
  __syncthreads();

  // coalesced b128 write-out of the staged [128 pt][128 ch] f16 tile
  _Float16* h4row = h4out + ((size_t)b * rowsPerBatch + tileBase) * 128;
  for (int i = 0; i < 8; ++i) {
    int c = t + 256 * i;                 // 2048 chunks of 8 f16
    int row = c >> 4, col = (c & 15) * 8;
    *(v8h*)&h4row[(size_t)row * 128 + col] = *(const v8h*)&stg[row * 128 + col];
  }
}

// ------------- conv5 (128 -> 1024) fused with per-tile max/argmax ----------
// Block tile: 128 points x 1024 channels. A-tile staged global->LDS with the
// gfx1250 async-to-LDS unit (ASYNCcnt), stride 136 f16 so 16 A-rows span all
// 64 LDS banks exactly once per ds_load_b128.
__global__ __launch_bounds__(256) void conv5_max_kernel(
    const _Float16* __restrict__ h4, const _Float16* __restrict__ W5,
    const float* __restrict__ b5, float* __restrict__ pmax,
    int* __restrict__ pidx) {
  __shared__ __align__(16) _Float16 sh[128 * 136];
  const int b = blockIdx.y, tile = blockIdx.x, t = threadIdx.x;
  const size_t gbase = ((size_t)b * NPT + tile * 128) * 128;
  // async copy of the 32 KB A tile: 8 x b128 per lane, tracked on ASYNCcnt
  for (int i = 0; i < 8; ++i) {
    int c = t + 256 * i;                  // 2048 chunks of 8 f16
    int row = c >> 4, col = (c & 15) * 8;
    unsigned ldsOff = (unsigned)(size_t)&sh[row * 136 + col];   // flat->LDS trunc
    const _Float16* gp = &h4[gbase + (size_t)row * 128 + col];
    asm volatile("global_load_async_to_lds_b128 %0, %1, off"
                 :: "v"(ldsOff), "v"(gp) : "memory");
  }
  asm volatile("s_wait_asynccnt 0" ::: "memory");
  __syncthreads();
  const int lane = t & 31, w = t >> 5, lrow = lane & 15, hi = lane >> 4;
  for (int cc = 0; cc < 8; ++cc) {
    int ch = (cc * 8 + w) * 16 + lrow;
    if (cc < 7)  // warm L2/WGP$ for next channel block of W5
      __builtin_prefetch(W5 + ((size_t)((cc + 1) * 8 + w) * 16 + lrow) * 128, 0, 1);
    v16h B0 = loadB(W5, ch, 128, 0,  hi);
    v16h B1 = loadB(W5, ch, 128, 32, hi);
    v16h B2 = loadB(W5, ch, 128, 64, hi);
    v16h B3 = loadB(W5, ch, 128, 96, hi);
    float bias = b5[ch];
    float best = -3.4e38f; int bidx = 0;
    for (int ps = 0; ps < 8; ++ps) {
      v8f acc = {};
      acc = wmma16(loadA(sh, ps * 16 + lrow, 136, 0,  hi), B0, acc);
      acc = wmma16(loadA(sh, ps * 16 + lrow, 136, 32, hi), B1, acc);
      acc = wmma16(loadA(sh, ps * 16 + lrow, 136, 64, hi), B2, acc);
      acc = wmma16(loadA(sh, ps * 16 + lrow, 136, 96, hi), B3, acc);
      for (int r = 0; r < 8; ++r) {          // points ascending per lane
        float vv = fmaxf(acc[r] + bias, 0.f);
        int pt = tile * 128 + ps * 16 + r + hi * 8;
        if (vv > best) { best = vv; bidx = pt; }   // first-occurrence
      }
    }
    float om = __shfl_xor(best, 16, 32);
    int   oi = __shfl_xor(bidx, 16, 32);
    if (om > best || (om == best && oi < bidx)) { best = om; bidx = oi; }
    if (lane < 16) {
      size_t o = ((size_t)b * EMB + ch) * 64 + tile;
      pmax[o] = best; pidx[o] = bidx;
    }
  }
}

__global__ void reduce_argmax_kernel(const float* __restrict__ pmax,
                                     const int* __restrict__ pidx,
                                     int* __restrict__ imp) {
  int i = blockIdx.x * blockDim.x + threadIdx.x;
  if (i >= NB * EMB) return;
  const float* pm = pmax + (size_t)i * 64;
  const int*   pi = pidx + (size_t)i * 64;
  float best = pm[0]; int bi = pi[0];
  for (int tt = 1; tt < 64; ++tt) {
    float m = pm[tt];
    if (m > best || (m == best && pi[tt] < bi)) { best = m; bi = pi[tt]; }
  }
  imp[i] = bi;
}

// -------------------- bincount + "first 600 empty bins" --------------------
__global__ void zero_counts_kernel(int* __restrict__ counts, int n) {
  int i = blockIdx.x * blockDim.x + threadIdx.x;
  if (i < n) counts[i] = 0;
}
__global__ void bincount_kernel(const int* __restrict__ imp, int* __restrict__ counts) {
  int i = blockIdx.x * blockDim.x + threadIdx.x;
  if (i >= NB * EMB) return;
  atomicAdd(&counts[(i >> 10) * NPT + imp[i]], 1);
}
// Stable ascending argsort of counts, take 600: since <=1024 bins are nonzero,
// this is exactly the first 600 indices with count==0. One wave per batch.
__global__ void select_kernel(const int* __restrict__ counts, int* __restrict__ sel) {
  int b = blockIdx.x, lane = threadIdx.x;
  int total = 0;
  for (int base = 0; base < NPT && total < KSEL; base += 32) {
    int i = base + lane;
    int zero = (counts[b * NPT + i] == 0) ? 1 : 0;
    unsigned mm = (unsigned)__ballot(zero);
    int pos = total + __popc(mm & ((1u << lane) - 1u));
    if (zero && pos < KSEL) sel[b * NPT2 + pos] = i;
    total += __popc(mm);
  }
}

// --------------- pass-2 conv5 + masked max-pool (one block/batch) ----------
__global__ __launch_bounds__(256) void conv5_pool2_kernel(
    const _Float16* __restrict__ h4b, const _Float16* __restrict__ W5,
    const float* __restrict__ b5, _Float16* __restrict__ pooledh) {
  const int b = blockIdx.x, t = threadIdx.x;
  const _Float16* A = h4b + (size_t)b * NPT2 * 128;
  const int lane = t & 31, w = t >> 5, lrow = lane & 15, hi = lane >> 4;
  for (int cc = 0; cc < 8; ++cc) {
    int ch = (cc * 8 + w) * 16 + lrow;
    v16h B0 = loadB(W5, ch, 128, 0,  hi);
    v16h B1 = loadB(W5, ch, 128, 32, hi);
    v16h B2 = loadB(W5, ch, 128, 64, hi);
    v16h B3 = loadB(W5, ch, 128, 96, hi);
    float bias = b5[ch];
    float best = -3.4e38f;
    for (int ps = 0; ps < NPT2 / 16; ++ps) {
      v8f acc = {};
      acc = wmma16(loadA(A, ps * 16 + lrow, 128, 0,  hi), B0, acc);
      acc = wmma16(loadA(A, ps * 16 + lrow, 128, 32, hi), B1, acc);
      acc = wmma16(loadA(A, ps * 16 + lrow, 128, 64, hi), B2, acc);
      acc = wmma16(loadA(A, ps * 16 + lrow, 128, 96, hi), B3, acc);
      for (int r = 0; r < 8; ++r) {
        int pt = ps * 16 + r + hi * 8;
        if (pt < KSEL) best = fmaxf(best, fmaxf(acc[r] + bias, 0.f));
      }
    }
    best = fmaxf(best, __shfl_xor(best, 16, 32));
    if (lane < 16) pooledh[b * EMB + ch] = (_Float16)best;
  }
}

// ------------------------ FC head: M = 16 batch rows -----------------------
__global__ __launch_bounds__(256) void fc_head_kernel(
    const _Float16* __restrict__ pooledh, const _Float16* __restrict__ L1e,
    const float* __restrict__ b6e, const _Float16* __restrict__ L2e,
    const float* __restrict__ L2b, float* __restrict__ out) {
  __shared__ __align__(16) _Float16 h6[16 * 520];
  const int t = threadIdx.x, lane = t & 31, w = t >> 5, lrow = lane & 15,
            hi = lane >> 4;
  for (int i = 0; i < 4; ++i) {           // FC1: [16,512] = [16,1024]x[1024,512]
    int ch = (w * 4 + i) * 16 + lrow;
    v8f acc = {};
    for (int kk = 0; kk < 32; ++kk) {
      v16h a = loadA(pooledh, lrow, 1024, kk * 32, hi);   // row = batch
      v16h bm = loadB(L1e, ch, 1024, kk * 32, hi);
      acc = wmma16(a, bm, acc);
    }
    float bias = b6e[ch];
    for (int r = 0; r < 8; ++r)
      h6[(r + hi * 8) * 520 + ch] = (_Float16)fmaxf(acc[r] + bias, 0.f);
  }
  __syncthreads();
  if (w < 3) {                            // FC2: [16,40(+pad 48)] = [16,512]x[512,48]
    int ch = w * 16 + lrow;
    v8f acc = {};
    for (int kk = 0; kk < 16; ++kk) {
      v16h a = loadA(h6, lrow, 520, kk * 32, hi);
      v16h bm = loadB(L2e, ch, 512, kk * 32, hi);
      acc = wmma16(a, bm, acc);
    }
    if (ch < 40) {
      float bias = L2b[ch];
      for (int r = 0; r < 8; ++r) out[(r + hi * 8) * 40 + ch] = acc[r] + bias;
    }
  }
}

// --------------------------------- host ------------------------------------
extern "C" void kernel_launch(void* const* d_in, const int* in_sizes, int n_in,
                              void* d_out, int out_size, void* d_ws, size_t ws_size,
                              hipStream_t stream) {
  (void)in_sizes; (void)n_in; (void)out_size; (void)ws_size;
  const float* x = (const float*)d_in[0];
  const float *W[5], *g[5], *bb[5], *mm[5], *vv[5];
  for (int i = 0; i < 5; ++i) {
    W[i] = (const float*)d_in[1 + i * 5];  g[i]  = (const float*)d_in[2 + i * 5];
    bb[i] = (const float*)d_in[3 + i * 5]; mm[i] = (const float*)d_in[4 + i * 5];
    vv[i] = (const float*)d_in[5 + i * 5];
  }
  const float* L1w = (const float*)d_in[26];
  const float* g6 = (const float*)d_in[27], *b6 = (const float*)d_in[28];
  const float* m6 = (const float*)d_in[29], *v6 = (const float*)d_in[30];
  const float* L2w = (const float*)d_in[31], *L2b = (const float*)d_in[32];

  char* p = (char*)d_ws;
  auto carve = [&](size_t bytes) -> void* {
    void* r = (void*)p; p += (bytes + 255) & ~(size_t)255; return r;
  };
  float*     W1e = (float*)carve(64 * 4 * 4);
  float*     b1e = (float*)carve(64 * 4);
  _Float16*  W2e = (_Float16*)carve(64 * 64 * 2);
  float*     b2e = (float*)carve(64 * 4);
  _Float16*  W3e = (_Float16*)carve(64 * 64 * 2);
  float*     b3e = (float*)carve(64 * 4);
  _Float16*  W4e = (_Float16*)carve(128 * 64 * 2);
  float*     b4e = (float*)carve(128 * 4);
  _Float16*  W5e = (_Float16*)carve((size_t)EMB * 128 * 2);
  float*     b5e = (float*)carve(EMB * 4);
  _Float16*  L1e = (_Float16*)carve((size_t)512 * EMB * 2);
  float*     b6e = (float*)carve(512 * 4);
  _Float16*  L2e = (_Float16*)carve((size_t)48 * 512 * 2);
  _Float16*  h4  = (_Float16*)carve((size_t)NB * NPT * 128 * 2);   // 32 MB
  float*     pmax = (float*)carve((size_t)NB * EMB * 64 * 4);      // 4 MB
  int*       pidx = (int*)carve((size_t)NB * EMB * 64 * 4);        // 4 MB
  int*       imp  = (int*)carve((size_t)NB * EMB * 4);
  int*       counts = (int*)carve((size_t)NB * NPT * 4);
  int*       sel  = (int*)carve((size_t)NB * NPT2 * 4);
  _Float16*  h4b  = (_Float16*)carve((size_t)NB * NPT2 * 128 * 2); // 2.5 MB
  _Float16*  pooledh = (_Float16*)carve((size_t)NB * EMB * 2);

  // 1) fold BN into weights (f16) once per launch
  fold_w1<<<1, 64, 0, stream>>>(W[0], g[0], bb[0], mm[0], vv[0], W1e, b1e);
  fold_bn_f16<<<16, 256, 0, stream>>>(W[1], g[1], bb[1], mm[1], vv[1], W2e, b2e, 64, 64);
  fold_bn_f16<<<16, 256, 0, stream>>>(W[2], g[2], bb[2], mm[2], vv[2], W3e, b3e, 64, 64);
  fold_bn_f16<<<32, 256, 0, stream>>>(W[3], g[3], bb[3], mm[3], vv[3], W4e, b4e, 128, 64);
  fold_bn_f16<<<512, 256, 0, stream>>>(W[4], g[4], bb[4], mm[4], vv[4], W5e, b5e, EMB, 128);
  fold_bn_f16<<<2048, 256, 0, stream>>>(L1w, g6, b6, m6, v6, L1e, b6e, 512, EMB);
  pad_l2<<<96, 256, 0, stream>>>(L2w, L2e);

  // 2) pass 1: conv1-4 -> h4 (f16), fused conv5 + max/argmax
  front_kernel<<<dim3(NPT / 128, NB), 256, 0, stream>>>(
      x, nullptr, W1e, b1e, W2e, b2e, W3e, b3e, W4e, b4e, h4, NPT, NPT, 0);
  conv5_max_kernel<<<dim3(NPT / 128, NB), 256, 0, stream>>>(h4, W5e, b5e, pmax, pidx);
  reduce_argmax_kernel<<<(NB * EMB) / 256, 256, 0, stream>>>(pmax, pidx, imp);

  // 3) bincount + select first 600 zero-count bins (== stable argsort take-600)
  zero_counts_kernel<<<(NB * NPT) / 256, 256, 0, stream>>>(counts, NB * NPT);
  bincount_kernel<<<(NB * EMB) / 256, 256, 0, stream>>>(imp, counts);
  select_kernel<<<NB, 32, 0, stream>>>(counts, sel);

  // 4) pass 2 on gathered 600 points (padded to 640, masked at the pool)
  front_kernel<<<dim3(NPT2 / 128, NB), 256, 0, stream>>>(
      x, sel, W1e, b1e, W2e, b2e, W3e, b3e, W4e, b4e, h4b, NPT2, KSEL, 1);
  conv5_pool2_kernel<<<NB, 256, 0, stream>>>(h4b, W5e, b5e, pooledh);
  fc_head_kernel<<<1, 256, 0, stream>>>(pooledh, L1e, b6e, L2e, L2b, (float*)d_out);
}
